// LSTMTransform_23201413333433
// MI455X (gfx1250) — matrix operand order repaired and checked by reading, not compiled, not run on hardware
//
#include <hip/hip_runtime.h>
#include <hip/hip_bf16.h>

typedef __attribute__((ext_vector_type(16))) _Float16 v16h;
typedef __attribute__((ext_vector_type(8)))  _Float16 v8h;
typedef __attribute__((ext_vector_type(8)))  float    v8f;

#define TT 4096
#define HH 256

static __device__ __forceinline__ float sigf(float x) { return 1.0f / (1.0f + __expf(-x)); }

// load 8 consecutive f32 and convert to v8h (two 16B vector loads)
static __device__ __forceinline__ v8h ld8h(const float* __restrict__ p) {
    float4 u = *(const float4*)p;
    float4 v = *(const float4*)(p + 4);
    v8h r;
    r[0] = (_Float16)u.x; r[1] = (_Float16)u.y; r[2] = (_Float16)u.z; r[3] = (_Float16)u.w;
    r[4] = (_Float16)v.x; r[5] = (_Float16)v.y; r[6] = (_Float16)v.z; r[7] = (_Float16)v.w;
    return r;
}
static __device__ __forceinline__ v16h combine(v8h lo, v8h hi) {
    v16h a;
#pragma unroll
    for (int q = 0; q < 8; ++q) { a[q] = lo[q]; a[q + 8] = hi[q]; }
    return a;
}

// C[M,N] = act(A[M,K] @ W[N,K]^T + bias[N]); 1 wave = 16x16 tile, block = 4 waves = 16x64.
__global__ __launch_bounds__(128) void wmma_gemm_bias(
    const float* __restrict__ A, int lda,
    const float* __restrict__ W, int ldw,
    const float* __restrict__ bias,
    float* __restrict__ C, int ldc, int K, int relu)
{
    const int lane = threadIdx.x & 31, w = threadIdx.x >> 5;
    const int half = lane >> 4, l16 = lane & 15;
    const int m0 = blockIdx.x * 16;
    const int n0 = blockIdx.y * 64 + w * 16;
    const float* arow = A + (long)(m0 + l16) * lda;
    const float* wrow = W + (long)(n0 + l16) * ldw;
    v8f acc = {};
    for (int kb = 0; kb < K; kb += 32) {
        // A 16x32 f16: lane holds row m0+l16, K [kb + half*16, +16)
        v16h a = combine(ld8h(arow + kb + half * 16), ld8h(arow + kb + half * 16 + 8));
        // B 32x16 f16: lane holds col n0+l16, K [kb + half*16, +16)
        v16h b = combine(ld8h(wrow + kb + half * 16), ld8h(wrow + kb + half * 16 + 8));
        acc = __builtin_amdgcn_wmma_f32_16x16x32_f16(false, a, false, b, (short)0, acc, false, false);
    }
#pragma unroll
    for (int r = 0; r < 8; ++r) {
        const int row = m0 + r + 8 * half;
        const int col = n0 + l16;
        float v = acc[r] + bias[col];
        if (relu) v = fmaxf(v, 0.0f);
        C[(long)row * ldc + col] = v;
    }
}

// Out[i,j] = 1 - max( (e_i . e_j) / (nrm[i]*nrm[j]), 1e-6 )
__global__ __launch_bounds__(128) void wmma_gram(
    const float* __restrict__ E, const float* __restrict__ nrm,
    float* __restrict__ Out, int N, int K)
{
    const int lane = threadIdx.x & 31, w = threadIdx.x >> 5;
    const int half = lane >> 4, l16 = lane & 15;
    const int m0 = blockIdx.x * 16;
    const int n0 = blockIdx.y * 64 + w * 16;
    const float* arow = E + (long)(m0 + l16) * K;
    const float* wrow = E + (long)(n0 + l16) * K;
    v8f acc = {};
    for (int kb = 0; kb < K; kb += 32) {
        v16h a = combine(ld8h(arow + kb + half * 16), ld8h(arow + kb + half * 16 + 8));
        v16h b = combine(ld8h(wrow + kb + half * 16), ld8h(wrow + kb + half * 16 + 8));
        acc = __builtin_amdgcn_wmma_f32_16x16x32_f16(false, a, false, b, (short)0, acc, false, false);
    }
#pragma unroll
    for (int r = 0; r < 8; ++r) {
        const int i = m0 + r + 8 * half;
        const int j = n0 + l16;
        float cs = acc[r] / (nrm[i] * nrm[j]);
        cs = fmaxf(cs, 1e-6f);
        Out[(long)i * N + j] = 1.0f - cs;
    }
}

__global__ __launch_bounds__(256) void row_norms(const float* __restrict__ E,
                                                 float* __restrict__ nrm, int K)
{
    const int row = blockIdx.x * 8 + (threadIdx.x >> 5);
    const int lane = threadIdx.x & 31;
    const float* p = E + (long)row * K;
    float s = 0.0f;
    for (int k = lane; k < K; k += 32) { float v = p[k]; s += v * v; }
#pragma unroll
    for (int off = 16; off; off >>= 1) s += __shfl_xor(s, off, 32);
    if (lane == 0) nrm[row] = sqrtf(s);
}

// One workgroup per direction (blockIdx.x = 0 fwd, 1 bwd), 32 waves.
// Whh (1024x256) is held persistently in VGPRs as f16 B-operands:
// wave w owns gate outputs [w*32, w*32+32) -> 2 N-tiles x 8 K-steps = 16 v16h regs.
__global__ __launch_bounds__(1024) void lstm_scan(
    const float* __restrict__ xs_f, const float* __restrict__ xs_b,
    const float* __restrict__ whh_f, const float* __restrict__ whh_b,
    float* __restrict__ hout /* T x 512 */)
{
    const int dir = blockIdx.x;
    const float* xs  = dir ? xs_b  : xs_f;
    const float* whh = dir ? whh_b : whh_f;
    const int colofs = dir ? HH : 0;

    __shared__ __align__(16) _Float16 hsh[HH];
    __shared__ float zsh[4 * HH];

    const int tid = threadIdx.x;
    const int lane = tid & 31, wid = tid >> 5;
    const int half = lane >> 4, l16 = lane & 15;

    // ---- persistent weight preload (B-operand layout, f16) ----
    v16h breg[2][8];
#pragma unroll
    for (int t2 = 0; t2 < 2; ++t2) {
        const int n0 = wid * 32 + t2 * 16;
        const float* wr = whh + (long)(n0 + l16) * HH;
#pragma unroll
        for (int ks = 0; ks < 8; ++ks) {
            const float* p = wr + ks * 32 + half * 16;
            breg[t2][ks] = combine(ld8h(p), ld8h(p + 8));
        }
    }

    if (tid < HH) hsh[tid] = (_Float16)0.0f;
    float c = 0.0f;
    __syncthreads();

    for (int t = 0; t < TT; ++t) {
        const int tt = dir ? (TT - 1 - t) : t;
        v8f acc0 = {}; v8f acc1 = {};
#pragma unroll
        for (int ks = 0; ks < 8; ++ks) {
            // A = h replicated to all 16 rows; per lane two 8-elem LDS vector loads
            const _Float16* hp = hsh + ks * 32 + half * 16;
            v16h a = combine(*(const v8h*)hp, *(const v8h*)(hp + 8));
            acc0 = __builtin_amdgcn_wmma_f32_16x16x32_f16(false, a, false, breg[0][ks], (short)0, acc0, false, false);
            acc1 = __builtin_amdgcn_wmma_f32_16x16x32_f16(false, a, false, breg[1][ks], (short)0, acc1, false, false);
        }
        if (half == 0) { // D row 0: (M=0, N=l16)
            zsh[wid * 32 + l16]      = acc0[0];
            zsh[wid * 32 + 16 + l16] = acc1[0];
        }
        __syncthreads();
        if (tid < HH) {
            const float* xr = xs + (long)tt * (4 * HH);
            const float zi = xr[tid]          + zsh[tid];
            const float zf = xr[HH + tid]     + zsh[HH + tid];
            const float zg = xr[2 * HH + tid] + zsh[2 * HH + tid];
            const float zo = xr[3 * HH + tid] + zsh[3 * HH + tid];
            c = sigf(zf) * c + sigf(zi) * tanhf(zg);
            const float h = sigf(zo) * tanhf(c);
            hout[(long)tt * (2 * HH) + colofs + tid] = h;
            hsh[tid] = (_Float16)h;
        }
        __syncthreads();
    }
}

extern "C" void kernel_launch(void* const* d_in, const int* in_sizes, int n_in,
                              void* d_out, int out_size, void* d_ws, size_t ws_size,
                              hipStream_t stream)
{
    const float* x       = (const float*)d_in[0];
    const float* wih_l0f = (const float*)d_in[1];
    const float* whh_l0f = (const float*)d_in[2];
    const float* b_l0f   = (const float*)d_in[3];
    const float* wih_l0b = (const float*)d_in[4];
    const float* whh_l0b = (const float*)d_in[5];
    const float* b_l0b   = (const float*)d_in[6];
    const float* wih_l1f = (const float*)d_in[7];
    const float* whh_l1f = (const float*)d_in[8];
    const float* b_l1f   = (const float*)d_in[9];
    const float* wih_l1b = (const float*)d_in[10];
    const float* whh_l1b = (const float*)d_in[11];
    const float* b_l1b   = (const float*)d_in[12];
    const float* fc1_w   = (const float*)d_in[13];
    const float* fc1_b   = (const float*)d_in[14];
    const float* fc2_w   = (const float*)d_in[15];
    const float* fc2_b   = (const float*)d_in[16];
    float* out = (float*)d_out;

    float* xsf = (float*)d_ws;                    // 4096*1024
    float* xsb = xsf + (size_t)TT * 1024;         // 4096*1024
    float* h1  = xsb + (size_t)TT * 1024;         // 4096*512
    float* h2  = h1  + (size_t)TT * 512;          // 4096*512
    float* zz  = h2  + (size_t)TT * 512;          // 4096*256
    float* ee  = zz  + (size_t)TT * 256;          // 4096*256
    float* nrm = ee  + (size_t)TT * 256;          // 4096

    const dim3 blk(128);

    // layer 0: xs = x @ Wih^T + b  (K=128, N=1024)
    wmma_gemm_bias<<<dim3(TT / 16, 1024 / 64), blk, 0, stream>>>(x, 128, wih_l0f, 128, b_l0f, xsf, 1024, 128, 0);
    wmma_gemm_bias<<<dim3(TT / 16, 1024 / 64), blk, 0, stream>>>(x, 128, wih_l0b, 128, b_l0b, xsb, 1024, 128, 0);
    lstm_scan<<<2, 1024, 0, stream>>>(xsf, xsb, whh_l0f, whh_l0b, h1);

    // layer 1: xs = h1 @ Wih^T + b  (K=512, N=1024)
    wmma_gemm_bias<<<dim3(TT / 16, 1024 / 64), blk, 0, stream>>>(h1, 512, wih_l1f, 512, b_l1f, xsf, 1024, 512, 0);
    wmma_gemm_bias<<<dim3(TT / 16, 1024 / 64), blk, 0, stream>>>(h1, 512, wih_l1b, 512, b_l1b, xsb, 1024, 512, 0);
    lstm_scan<<<2, 1024, 0, stream>>>(xsf, xsb, whh_l1f, whh_l1b, h2);

    // FC1 (ReLU), FC2
    wmma_gemm_bias<<<dim3(TT / 16, 256 / 64), blk, 0, stream>>>(h2, 512, fc1_w, 512, fc1_b, zz, 256, 512, 1);
    wmma_gemm_bias<<<dim3(TT / 16, 256 / 64), blk, 0, stream>>>(zz, 256, fc2_w, 256, fc2_b, ee, 256, 256, 0);

    // norms + fused cosine-distance Gram
    row_norms<<<TT / 8, 256, 0, stream>>>(ee, nrm, 256);
    wmma_gram<<<dim3(TT / 16, TT / 64), blk, 0, stream>>>(ee, nrm, out, TT, 256);
}
